// RelationModel_43774306680840
// MI455X (gfx1250) — matrix operand order repaired and checked by reading, not compiled, not run on hardware
//
#include <hip/hip_runtime.h>
#include <hip/hip_bf16.h>
#include <math.h>

// ---------------- problem constants ----------------
#define BATCH 8
#define LPROG 12
#define CDIM  4096      // MAX_CONCEPTS
#define ODIM  128       // MAX_OBJECTS
#define NDIM  4224      // CDIM + ODIM
#define EDIM  64
#define ADIM  64
#define HDIM  256

typedef __attribute__((ext_vector_type(8)))  float   v8f;
typedef __attribute__((ext_vector_type(16))) __bf16  v16bf;

union BF16x16 { v16bf v; unsigned int u[8]; };

// pack two fp32 into a bf16 pair (lo in [15:0], hi in [31:16]) with ONE v_perm_b32
__device__ __forceinline__ unsigned pack2(float lo, float hi) {
    return __builtin_amdgcn_perm(__float_as_uint(hi), __float_as_uint(lo), 0x07060302u);
}

__device__ __forceinline__ v8f wmma_bf16(const BF16x16& a, const BF16x16& b, v8f c) {
    return __builtin_amdgcn_wmma_f32_16x16x32_bf16(false, a.v, false, b.v, (short)0, c,
                                                   false, false);
}

// A fragment (16x32 bf16): lane holds row m=lane&15; K pairs
// kk(v) = k0 + (v<4?0:16) + 8*hh + 2*(v&3)  -> two contiguous 8-float runs.
__device__ __forceinline__ void load_a(BF16x16& a, const float* rowptr, int k0, int hh) {
    const float* p = rowptr + k0 + 8 * hh;
    float4 f0 = *(const float4*)(p);
    float4 f1 = *(const float4*)(p + 4);
    float4 f2 = *(const float4*)(p + 16);
    float4 f3 = *(const float4*)(p + 20);
    a.u[0] = pack2(f0.x, f0.y); a.u[1] = pack2(f0.z, f0.w);
    a.u[2] = pack2(f1.x, f1.y); a.u[3] = pack2(f1.z, f1.w);
    a.u[4] = pack2(f2.x, f2.y); a.u[5] = pack2(f2.z, f2.w);
    a.u[6] = pack2(f3.x, f3.y); a.u[7] = pack2(f3.z, f3.w);
}

// Same with fused bias-add + ReLU (hidden activation of the axon MLP).
__device__ __forceinline__ void load_a_relu(BF16x16& a, const float* rowptr,
                                            const float* hb, int k0, int hh) {
    const float* p = rowptr + k0 + 8 * hh;
    const float* q = hb + k0 + 8 * hh;
    float4 f0 = *(const float4*)(p);       float4 g0 = *(const float4*)(q);
    float4 f1 = *(const float4*)(p + 4);   float4 g1 = *(const float4*)(q + 4);
    float4 f2 = *(const float4*)(p + 16);  float4 g2 = *(const float4*)(q + 16);
    float4 f3 = *(const float4*)(p + 20);  float4 g3 = *(const float4*)(q + 20);
#define RL(x, y) fmaxf((x) + (y), 0.0f)
    a.u[0] = pack2(RL(f0.x,g0.x), RL(f0.y,g0.y)); a.u[1] = pack2(RL(f0.z,g0.z), RL(f0.w,g0.w));
    a.u[2] = pack2(RL(f1.x,g1.x), RL(f1.y,g1.y)); a.u[3] = pack2(RL(f1.z,g1.z), RL(f1.w,g1.w));
    a.u[4] = pack2(RL(f2.x,g2.x), RL(f2.y,g2.y)); a.u[5] = pack2(RL(f2.z,g2.z), RL(f2.w,g2.w));
    a.u[6] = pack2(RL(f3.x,g3.x), RL(f3.y,g3.y)); a.u[7] = pack2(RL(f3.z,g3.z), RL(f3.w,g3.w));
#undef RL
}

// A fragment from a pre-packed bf16 matrix (u32 = bf16 pair, pairs along K).
__device__ __forceinline__ void load_a_packed(BF16x16& a, const unsigned* rowu, int k0, int hh) {
    const uint4* p = (const uint4*)(rowu + ((k0 + 8 * hh) >> 1));
    uint4 w0 = p[0];        // K = k0+8h .. +7
    uint4 w1 = p[2];        // K = k0+16+8h .. +7  (+8 u32)
    a.u[0] = w0.x; a.u[1] = w0.y; a.u[2] = w0.z; a.u[3] = w0.w;
    a.u[4] = w1.x; a.u[5] = w1.y; a.u[6] = w1.z; a.u[7] = w1.w;
}

// B fragment (32x16 bf16) from fp32: lane <-> N col n0+(lane&15), VGPR v <-> K pair kk(v) as in A.
__device__ __forceinline__ void load_b(BF16x16& b, const float* B, int ldb,
                                       int k0, int n0, int lane) {
    const float* col = B + n0 + (lane & 15);
    int hh = lane >> 4;
#pragma unroll
    for (int v = 0; v < 8; ++v) {
        int k = k0 + (v < 4 ? 0 : 16) + 8 * hh + 2 * (v & 3);
        b.u[v] = pack2(col[(size_t)k * ldb], col[(size_t)(k + 1) * ldb]);
    }
}

// B fragment from a pre-packed bf16 matrix (ldbu in u32 units = cols/2); lane <-> N col, VGPR v <-> K pair kk(v).
__device__ __forceinline__ void load_b_packed(BF16x16& b, const unsigned* Bp, int ldbu,
                                              int k0, int n0, int lane) {
    const unsigned short* col = (const unsigned short*)Bp + n0 + (lane & 15);
    size_t ld = (size_t)ldbu * 2;
    int hh = lane >> 4;
#pragma unroll
    for (int v = 0; v < 8; ++v) {
        int k = k0 + (v < 4 ? 0 : 16) + 8 * hh + 2 * (v & 3);
        b.u[v] = (unsigned)col[(size_t)k * ld] | ((unsigned)col[(size_t)(k + 1) * ld] << 16);
    }
}

// ---------------- pack a row-major fp32 matrix into bf16-pair dwords ----------
__global__ void pack_pairs(const float* __restrict__ in, unsigned* __restrict__ out) {
    int i = blockIdx.x * blockDim.x + threadIdx.x;   // total = rows*cols/2 dwords
    float2 f = *(const float2*)(in + 2 * i);
    out[i] = pack2(f.x, f.y);
}

// ---------------- K0: attention init ----------------
__global__ void init_att(const float* __restrict__ att_init, float* __restrict__ att) {
    int idx = blockIdx.x * blockDim.x + threadIdx.x;   // BATCH*NDIM*ADIM threads
    att[idx] = att_init[idx & (ADIM - 1)];
}

// ---------------- K1: object embeddings ----------------
__global__ void build_obj(const float* __restrict__ class_embIn, const float* __restrict__ class_embOut,
                          const float* __restrict__ attr_embIn,  const float* __restrict__ attr_embOut,
                          const int* __restrict__ gt_classes,    const int* __restrict__ gt_attrs,
                          float* __restrict__ objIn, float* __restrict__ objOut) {
    int idx = blockIdx.x * blockDim.x + threadIdx.x;   // BATCH*ODIM*EDIM threads
    int e = idx & 63;
    int bo = idx >> 6;                                  // b*128 + o
    int cls = gt_classes[bo];
    float si = class_embIn[cls * 64 + e];
    float so = class_embOut[cls * 64 + e];
#pragma unroll
    for (int p = 0; p < 8; ++p) {
        int at = gt_attrs[bo * 8 + p];
        si += attr_embIn[at * 64 + e];
        so += attr_embOut[at * 64 + e];
    }
    objIn[idx] = si;
    objOut[idx] = so;
}

// ---------------- K2: meta recurrence (state-independent) ----------------
__global__ void meta_scan(const float* __restrict__ concept_embOut, const float* __restrict__ meta_init,
                          const float* __restrict__ mw1, const float* __restrict__ mb1,
                          const float* __restrict__ mw2, const float* __restrict__ mb2,
                          const float* __restrict__ aw1, const float* __restrict__ ab1,
                          const int* __restrict__ ops,  const int* __restrict__ args,
                          float* __restrict__ meta_all, float* __restrict__ hbias_all) {
    __shared__ float meta[BATCH * 64];
    __shared__ float hid[BATCH * 256];
    int tid = threadIdx.x;
    for (int i = tid; i < BATCH * 64; i += 256) meta[i] = meta_init[i & 63];
    __syncthreads();
    for (int t = 0; t < LPROG; ++t) {
        for (int i = tid; i < BATCH * 256; i += 256) {
            int b = i >> 8, h = i & 255;
            const float* arg = concept_embOut + (size_t)args[b * LPROG + t] * 64;
            float s = mb1[h];
            for (int k = 0; k < 64; ++k) s += meta[b * 64 + k] * mw1[k * 256 + h];
            for (int k = 0; k < 64; ++k) s += arg[k] * mw1[(64 + k) * 256 + h];
            hid[i] = fmaxf(s, 0.0f);
        }
        __syncthreads();
        for (int i = tid; i < BATCH * 64; i += 256) {
            int b = i >> 6, a = i & 63;
            float s = mb2[a];
            for (int k = 0; k < 256; ++k) s += hid[b * 256 + k] * mw2[k * 64 + a];
            float m = (ops[b * LPROG + t] == 0) ? 1.0f : 0.0f;
            float nv = m * s + (1.0f - m) * meta[i];
            meta[i] = nv;
            meta_all[t * (BATCH * 64) + i] = nv;
        }
        __syncthreads();
        for (int i = tid; i < BATCH * 256; i += 256) {
            int b = i >> 8, h = i & 255;
            float s = ab1[h];
            for (int k = 0; k < 64; ++k) s += meta[b * 64 + k] * aw1[(64 + k) * 256 + h];
            hbias_all[t * (BATCH * 256) + i] = s;
        }
        __syncthreads();
    }
}

// ---------------- K3: proj[t][b][n] = thoughtOut . arg / E ----------------
__global__ void proj_kernel(const float* __restrict__ concept_embOut, const float* __restrict__ objOut,
                            const int* __restrict__ args, float* __restrict__ proj) {
    int idx = blockIdx.x * blockDim.x + threadIdx.x;   // LPROG*BATCH*NDIM threads
    int n = idx % NDIM;
    int rem = idx / NDIM;
    int b = rem % BATCH;
    int t = rem / BATCH;
    const float* arg = concept_embOut + (size_t)args[b * LPROG + t] * 64;
    const float* row = (n < CDIM) ? (concept_embOut + (size_t)n * 64)
                                  : (objOut + (size_t)(b * ODIM + (n - CDIM)) * 64);
    float s = 0.0f;
#pragma unroll 8
    for (int e = 0; e < 64; ++e) s += row[e] * arg[e];
    proj[idx] = s * (1.0f / 64.0f);
}

// ---------------- K4: H0 = thoughtOut @ axon_w1[:64,:]  (WMMA) ----------------
__global__ void h0_gemm(const float* __restrict__ A, const unsigned* __restrict__ w1p,
                        float* __restrict__ Out) {
    int wid = (blockIdx.x * blockDim.x + threadIdx.x) >> 5;
    int lane = threadIdx.x & 31;
    int hh = lane >> 4, ml = lane & 15;
    int nt = wid & 15;
    int mt = wid >> 4;
    const float* rowptr = A + (size_t)(mt * 16 + ml) * 64;
    v8f acc = {};
    BF16x16 af, bf;
#pragma unroll
    for (int c = 0; c < 2; ++c) {
        load_a(af, rowptr, c * 32, hh);
        load_b_packed(bf, w1p, 128, c * 32, nt * 16, lane);
        acc = wmma_bf16(af, bf, acc);
    }
#pragma unroll
    for (int v = 0; v < 8; ++v)
        Out[(size_t)(mt * 16 + v + 8 * hh) * 256 + nt * 16 + ml] = acc[v];
}

// ---------------- K5: axonT[b][e][n] (bf16 packed) = (relu(H0+hb) @ w2 + b2)ᵀ
// waves: b(8) x rt(264) x et(4)
__global__ void axon_kernel(const float* __restrict__ H0c, const float* __restrict__ H0o,
                            const float* __restrict__ hbias_t, const unsigned* __restrict__ w2p,
                            const float* __restrict__ b2, unsigned* __restrict__ axonT) {
    int wid = (blockIdx.x * blockDim.x + threadIdx.x) >> 5;
    int lane = threadIdx.x & 31;
    int hh = lane >> 4, ml = lane & 15;
    int et = wid & 3;
    int rt = (wid >> 2) % (NDIM / 16);
    int b  = wid / (4 * (NDIM / 16));
    int r  = rt * 16 + ml;
    const float* rowptr = (r < CDIM) ? (H0c + (size_t)r * 256)
                                     : (H0o + (size_t)(b * ODIM + (r - CDIM)) * 256);
    const float* hb = hbias_t + b * 256;
    v8f acc = {};
    BF16x16 af, bf;
#pragma unroll
    for (int c = 0; c < HDIM / 32; ++c) {       // 8 WMMAs over K=256
        load_a_relu(af, rowptr, hb, c * 32, hh);
        load_b_packed(bf, w2p, 32, c * 32, et * 16, lane);
        acc = wmma_bf16(af, bf, acc);
    }
    int e_col = et * 16 + ml;
    float bias = b2[e_col];
    // transposed, pre-packed bf16 store: lane owns 8 contiguous n at row e_col
    uint4 w;
    w.x = pack2(acc[0] + bias, acc[1] + bias);
    w.y = pack2(acc[2] + bias, acc[3] + bias);
    w.z = pack2(acc[4] + bias, acc[5] + bias);
    w.w = pack2(acc[6] + bias, acc[7] + bias);
    *(uint4*)(axonT + (size_t)(b * EDIM + e_col) * (NDIM / 2) + rt * 8 + 4 * hh) = w;
}

// ---------------- K6: M[b] = axonᵀ @ attention  (WMMA, K split + atomics) ----
// waves: seg(12) x b(8) x et(4) x at(4); each seg covers 352 of N
__global__ void mgemm_kernel(const unsigned* __restrict__ axonT, const float* __restrict__ att,
                             float* __restrict__ M) {
    int wid = (blockIdx.x * blockDim.x + threadIdx.x) >> 5;
    int lane = threadIdx.x & 31;
    int hh = lane >> 4, ml = lane & 15;
    int at = wid & 3;
    int et = (wid >> 2) & 3;
    int b  = (wid >> 4) & 7;
    int seg = wid >> 7;
    const unsigned* rowu = axonT + (size_t)(b * EDIM + et * 16 + ml) * (NDIM / 2);
    const float* attB = att + (size_t)b * NDIM * ADIM;
    v8f acc = {};
    BF16x16 af, bf;
    int kbase = seg * 352;
#pragma unroll
    for (int c = 0; c < 11; ++c) {
        int k0 = kbase + c * 32;
        load_a_packed(af, rowu, k0, hh);
        load_b(bf, attB, ADIM, k0, at * 16, lane);
        acc = wmma_bf16(af, bf, acc);
    }
#pragma unroll
    for (int v = 0; v < 8; ++v)
        atomicAdd(M + (size_t)(b * EDIM + et * 16 + v + 8 * hh) * ADIM + at * 16 + ml, acc[v]);
}

// ---------------- K7: attention += ins*att_insert + tr*(thoughtIn@M)/N  (WMMA)
// waves: b(8) x rt(264) x at(4);  Mp = pre-packed bf16 M (64x32 dwords per batch)
__global__ void update_kernel(const float* __restrict__ conceptIn, const float* __restrict__ objIn,
                              const unsigned* __restrict__ Mp, const float* __restrict__ meta_all,
                              const float* __restrict__ proj, const int* __restrict__ ops,
                              float* __restrict__ att, int t) {
    int wid = (blockIdx.x * blockDim.x + threadIdx.x) >> 5;
    int lane = threadIdx.x & 31;
    int hh = lane >> 4, ml = lane & 15;
    int at = wid & 3;
    int rt = (wid >> 2) % (NDIM / 16);
    int b  = wid / (4 * (NDIM / 16));
    int r0 = rt * 16 + ml;
    const float* rowptr = (r0 < CDIM) ? (conceptIn + (size_t)r0 * 64)
                                      : (objIn + (size_t)(b * ODIM + (r0 - CDIM)) * 64);
    const unsigned* Mb = Mp + (size_t)b * EDIM * (ADIM / 2);
    v8f acc = {};
    BF16x16 af, bf;
#pragma unroll
    for (int c = 0; c < 2; ++c) {               // K = EDIM = 64
        load_a(af, rowptr, c * 32, hh);
        load_b_packed(bf, Mb, ADIM / 2, c * 32, at * 16, lane);
        acc = wmma_bf16(af, bf, acc);
    }
    int op = ops[b * LPROG + t];
    float ins = (op == 1) ? 1.0f : 0.0f;
    float tr  = (op == 2) ? 1.0f : 0.0f;
    int a_col = at * 16 + ml;
    float mta = meta_all[(size_t)(t * BATCH + b) * ADIM + a_col];
    int rbase = rt * 16 + 8 * hh;
#pragma unroll
    for (int v = 0; v < 8; ++v) {
        int r = rbase + v;
        float p = proj[(size_t)(t * BATCH + b) * NDIM + r];
        float* dst = att + ((size_t)b * NDIM + r) * ADIM + a_col;
        *dst += ins * mta * p + tr * acc[v] * (1.0f / (float)NDIM);
    }
}

// ---------------- K8: output_length + log_softmax ----------------
__global__ void out_kernel(const float* __restrict__ att, float* __restrict__ out) {
    int b = blockIdx.x, tid = threadIdx.x;
    __shared__ float vals[NDIM];
    __shared__ float red[256];
    for (int n = tid; n < NDIM; n += 256) {
        const float* row = att + ((size_t)b * NDIM + n) * ADIM;
        float s = 0.0f;
#pragma unroll 8
        for (int a = 0; a < ADIM; ++a) { float x = row[a]; s += x * x; }
        vals[n] = s * (1.0f / (float)ADIM);
    }
    __syncthreads();
    float mx = -INFINITY;
    for (int n = tid; n < NDIM; n += 256) mx = fmaxf(mx, vals[n]);
    red[tid] = mx; __syncthreads();
    for (int s = 128; s > 0; s >>= 1) { if (tid < s) red[tid] = fmaxf(red[tid], red[tid + s]); __syncthreads(); }
    mx = red[0]; __syncthreads();
    float sm = 0.0f;
    for (int n = tid; n < NDIM; n += 256) sm += expf(vals[n] - mx);
    red[tid] = sm; __syncthreads();
    for (int s = 128; s > 0; s >>= 1) { if (tid < s) red[tid] += red[tid + s]; __syncthreads(); }
    float lse = logf(red[0]) + mx;
    for (int n = tid; n < NDIM; n += 256) out[b * NDIM + n] = vals[n] - lse;
}

// ---------------- host launch ----------------
extern "C" void kernel_launch(void* const* d_in, const int* in_sizes, int n_in,
                              void* d_out, int out_size, void* d_ws, size_t ws_size,
                              hipStream_t stream) {
    (void)in_sizes; (void)n_in; (void)out_size; (void)ws_size;
    const float* class_embIn   = (const float*)d_in[0];
    const float* class_embOut  = (const float*)d_in[1];
    const float* attr_embIn    = (const float*)d_in[2];
    const float* attr_embOut   = (const float*)d_in[3];
    const float* concept_embIn = (const float*)d_in[4];
    const float* concept_embOut= (const float*)d_in[5];
    const float* meta_init     = (const float*)d_in[6];
    /* d_in[7] object_init is unused by the reference */
    const float* attention_init= (const float*)d_in[8];
    const float* axon_w1 = (const float*)d_in[9];
    const float* axon_b1 = (const float*)d_in[10];
    const float* axon_w2 = (const float*)d_in[11];
    const float* axon_b2 = (const float*)d_in[12];
    const float* meta_w1 = (const float*)d_in[13];
    const float* meta_b1 = (const float*)d_in[14];
    const float* meta_w2 = (const float*)d_in[15];
    const float* meta_b2 = (const float*)d_in[16];
    const int* program_ops   = (const int*)d_in[17];
    const int* program_args  = (const int*)d_in[18];
    const int* gt_classes    = (const int*)d_in[19];
    const int* gt_attributes = (const int*)d_in[20];

    float* ws       = (float*)d_ws;
    float*    objIn    = ws;                          // 8*128*64        =   65536
    float*    objOut   = objIn    + 65536;            //                 =   65536
    float*    H0c      = objOut   + 65536;            // 4096*256        = 1048576
    float*    H0o      = H0c      + 1048576;          // 1024*256        =  262144
    float*    meta_all = H0o      + 262144;           // 12*8*64         =    6144
    float*    hbias    = meta_all + 6144;             // 12*8*256        =   24576
    float*    proj     = hbias    + 24576;            // 12*8*4224       =  405504
    unsigned* axonT    = (unsigned*)(proj + 405504);  // 8*64*2112 u32   = 1081344
    float*    Mws      = (float*)(axonT + 1081344);   // 8*64*64         =   32768
    unsigned* Mp       = (unsigned*)(Mws + 32768);    // 8*64*32 u32     =   16384
    unsigned* w1p      = Mp + 16384;                  // 64*128 u32      =    8192
    unsigned* w2p      = w1p + 8192;                  // 256*32 u32      =    8192
    float*    att      = (float*)(w2p + 8192);        // 8*4224*64       = 2162688
    // total ~5.0M dwords (~20 MB) -> L2-resident

    // one-time pre-packing of step-invariant B operands into bf16-pair layout
    pack_pairs<<<(64 * 256 / 2) / 256, 256, 0, stream>>>(axon_w1, w1p);   // w1[:64,:]
    pack_pairs<<<(256 * 64 / 2) / 256, 256, 0, stream>>>(axon_w2, w2p);

    init_att <<<(BATCH * NDIM * ADIM) / 256, 256, 0, stream>>>(attention_init, att);
    build_obj<<<(BATCH * ODIM * EDIM) / 256, 256, 0, stream>>>(
        class_embIn, class_embOut, attr_embIn, attr_embOut,
        gt_classes, gt_attributes, objIn, objOut);
    meta_scan<<<1, 256, 0, stream>>>(concept_embOut, meta_init,
        meta_w1, meta_b1, meta_w2, meta_b2, axon_w1, axon_b1,
        program_ops, program_args, meta_all, hbias);
    proj_kernel<<<(LPROG * BATCH * NDIM) / 256, 256, 0, stream>>>(
        concept_embOut, objOut, program_args, proj);
    h0_gemm<<<(CDIM / 16) * 16 / 8, 256, 0, stream>>>(concept_embOut, w1p, H0c);
    h0_gemm<<<((BATCH * ODIM) / 16) * 16 / 8, 256, 0, stream>>>(objOut, w1p, H0o);

    for (int t = 0; t < LPROG; ++t) {
        axon_kernel<<<(BATCH * (NDIM / 16) * 4) / 8, 256, 0, stream>>>(
            H0c, H0o, hbias + t * BATCH * HDIM, w2p, axon_b2, axonT);
        hipMemsetAsync(Mws, 0, (size_t)BATCH * EDIM * ADIM * sizeof(float), stream);
        mgemm_kernel<<<(12 * BATCH * 4 * 4) / 8, 256, 0, stream>>>(axonT, att, Mws);
        pack_pairs<<<(BATCH * EDIM * ADIM / 2) / 256, 256, 0, stream>>>(Mws, Mp);
        update_kernel<<<(BATCH * (NDIM / 16) * 4) / 8, 256, 0, stream>>>(
            concept_embIn, objIn, Mp, meta_all, proj, program_ops, att, t);
    }
    out_kernel<<<BATCH, 256, 0, stream>>>(att, (float*)d_out);
}
